// OpticalDepth_23364622090767
// MI455X (gfx1250) — matrix operand order repaired and checked by reading, not compiled, not run on hardware
//
#include <hip/hip_runtime.h>
#include <hip/hip_bf16.h>

typedef __attribute__((ext_vector_type(16))) _Float16 v16h;
typedef __attribute__((ext_vector_type(2)))  _Float16 v2h;
typedef __attribute__((ext_vector_type(8)))  float    v8f;

#define NNETS 76
#define NCH   30
#define H     64

// gas id per net: 29 h2o, 13 o3, 9 co2, 3 n2o, 9 ch4, 13 u
__constant__ unsigned char GASID[NNETS] = {
    0,0,0,0,0,0,0,0,0,0,0,0,0,0,0,0,0,0,0,0,0,0,0,0,0,0,0,0,0,
    1,1,1,1,1,1,1,1,1,1,1,1,1,
    2,2,2,2,2,2,2,2,2,
    3,3,3,
    4,4,4,4,4,4,4,4,4,
    5,5,5,5,5,5,5,5,5,5,5,5,5
};

// channel -> contributing nets (-1 padded), from the hand-coded tree
__constant__ int SEL[NCH][6] = {
    {0,29,42,51,54,63}, {0,29,42,51,54,63},
    {1,30,43,52,55,64}, {2,31,44,53,56,65},
    {3,57,-1,-1,-1,-1}, {4,58,-1,-1,-1,-1},
    {5,45,-1,-1,-1,-1}, {6,46,-1,-1,-1,-1},
    {7,59,-1,-1,-1,-1}, {8,60,-1,-1,-1,-1},
    {9,47,-1,-1,-1,-1}, {10,48,-1,-1,-1,-1},
    {11,61,-1,-1,-1,-1}, {12,62,-1,-1,-1,-1},
    {13,49,-1,-1,-1,-1}, {14,50,-1,-1,-1,-1},
    {15,66,-1,-1,-1,-1}, {16,67,-1,-1,-1,-1},
    {17,32,68,-1,-1,-1}, {18,33,69,-1,-1,-1},
    {19,34,70,-1,-1,-1}, {20,35,71,-1,-1,-1},
    {21,36,72,-1,-1,-1}, {22,37,73,-1,-1,-1},
    {23,-1,-1,-1,-1,-1}, {24,-1,-1,-1,-1,-1},
    {25,38,-1,-1,-1,-1}, {26,39,-1,-1,-1,-1},
    {27,40,74,-1,-1,-1}, {28,41,75,-1,-1,-1}
};

// One workgroup = (net, 256-row batch tile). 128 threads = 4 waves.
// Each wave owns 64 rows: layer1 in packed f16 VALU -> wave-private LDS,
// layer2 = 32 x v_wmma_f32_16x16x32_f16 with register-cached B fragments,
// layer3 fused from accumulators via shfl_xor butterfly.
__global__ __launch_bounds__(128)
void mlp_fused_kernel(const float* __restrict__ t_p,
                      const float* __restrict__ comp,
                      const float* __restrict__ W1, const float* __restrict__ b1,
                      const float* __restrict__ W2, const float* __restrict__ b2,
                      const float* __restrict__ W3, const float* __restrict__ b3,
                      float* __restrict__ tau_ws, int B) {
    __shared__ _Float16 sW2t[H * H];        // W2 transposed [col][k], f16 (8KB)
    __shared__ _Float16 sH1[4 * H * H];     // 4 wave-private 64x64 tiles (32KB)
    __shared__ v2h      sW1h0[H / 2], sW1h1[H / 2], sB1h[H / 2];
    __shared__ float    sB2[H], sW3[H];

    const int tid  = threadIdx.x;
    const int net  = blockIdx.y;
    const int row0 = blockIdx.x * 256;

    // ---- stage weights once per block (amortized over 256 rows)
    const float* W2n = W2 + net * (H * H);
    for (int idx = tid; idx < H * H; idx += 128) {
        int k = idx >> 6, j = idx & 63;
        sW2t[j * H + k] = (_Float16)W2n[idx];   // transpose: B-frags contiguous
    }
    if (tid < H / 2) {                           // W1/b1 as packed f16 pairs
        const float* W1n = W1 + net * 2 * H;
        const float* b1n = b1 + net * H;
        sW1h0[tid] = (v2h){(_Float16)W1n[2 * tid],     (_Float16)W1n[2 * tid + 1]};
        sW1h1[tid] = (v2h){(_Float16)W1n[H + 2 * tid], (_Float16)W1n[H + 2 * tid + 1]};
        sB1h[tid]  = (v2h){(_Float16)b1n[2 * tid],     (_Float16)b1n[2 * tid + 1]};
    }
    if (tid < H) { sB2[tid] = b2[net * H + tid];
                   sW3[tid] = W3[net * H + tid]; }
    __syncthreads();                            // only barrier in the kernel

    const int wave    = tid >> 5;
    const int lane    = tid & 31;
    const int half_id = lane >> 4;
    const int l16     = lane & 15;
    const int rbase   = row0 + wave * 64;       // this wave's 64 rows

    _Float16* sH1w = sH1 + wave * (H * H);      // wave-private region

    // ---- layer 1 (packed f16): each lane does 2 rows x 32 col-pairs
    // v_pk_fma_f16 / v_pk_max produce pairs already packed for b128 stores
    float4 tp4 = *(const float4*)(t_p + 2 * (rbase + lane * 2)); // coalesced
    const v2h zero2 = (v2h){(_Float16)0.f, (_Float16)0.f};
    #pragma unroll
    for (int rr = 0; rr < 2; ++rr) {
        v2h tp0h = (v2h)((_Float16)(rr ? tp4.z : tp4.x));   // splat
        v2h tp1h = (v2h)((_Float16)(rr ? tp4.w : tp4.y));   // splat
        _Float16* dst = sH1w + (lane * 2 + rr) * H;
        #pragma unroll
        for (int p0 = 0; p0 < H / 2; p0 += 4) {
            union { v2h h2[4]; uint4 u; } pk;
            #pragma unroll
            for (int pp = 0; pp < 4; ++pp) {
                int p = p0 + pp;
                v2h v = tp0h * sW1h0[p] + tp1h * sW1h1[p] + sB1h[p];
                pk.h2[pp] = __builtin_elementwise_max(v, zero2);   // relu
            }
            *(uint4*)(dst + 2 * p0) = pk.u;
        }
    }

    // ---- load all 8 B fragments into registers once (reused by 4 strips)
    // B layout: lane holds col N=l16+16t; lanes 0-15: K=0..15, 16-31: K=16..31
    v16h bfrag[2][4];
    #pragma unroll
    for (int kt = 0; kt < 2; ++kt)
        #pragma unroll
        for (int t = 0; t < 4; ++t) {
            const _Float16* bp = sW2t + (l16 + 16 * t) * H + kt * 32 + 16 * half_id;
            ((uint4*)&bfrag[kt][t])[0] = *(const uint4*)(bp);
            ((uint4*)&bfrag[kt][t])[1] = *(const uint4*)(bp + 8);
        }

    float bias_t[4], w3_t[4];
    #pragma unroll
    for (int t = 0; t < 4; ++t) {
        int c = l16 + 16 * t;
        bias_t[t] = sB2[c];
        w3_t[t]   = sW3[c];
    }
    const float b3n = b3[net];
    const int   g   = (int)GASID[net];

    // ---- 4 strips of 16 rows: 8 WMMAs each, then fused layer3 reduction
    #pragma unroll
    for (int strip = 0; strip < 4; ++strip) {
        v8f acc[4];
        #pragma unroll
        for (int t = 0; t < 4; ++t)
            acc[t] = (v8f){0.f,0.f,0.f,0.f,0.f,0.f,0.f,0.f};

        // A layout: lane holds row M=l16; lanes 0-15: K=0..15, 16-31: K=16..31 (+32kt)
        const _Float16* aRow = sH1w + (strip * 16 + l16) * H;
        #pragma unroll
        for (int kt = 0; kt < 2; ++kt) {
            v16h afrag;
            const _Float16* ap = aRow + kt * 32 + 16 * half_id;
            ((uint4*)&afrag)[0] = *(const uint4*)(ap);
            ((uint4*)&afrag)[1] = *(const uint4*)(ap + 8);
            #pragma unroll
            for (int t = 0; t < 4; ++t)
                acc[t] = __builtin_amdgcn_wmma_f32_16x16x32_f16(
                    false, afrag, false, bfrag[kt][t], (short)0, acc[t], false, false);
        }

        // layer3 partials: lane covers cols {l16, l16+16, l16+32, l16+48}
        // C layout: VGPR r -> row M = r + 8*half_id
        float p[8] = {0.f,0.f,0.f,0.f,0.f,0.f,0.f,0.f};
        #pragma unroll
        for (int t = 0; t < 4; ++t)
            #pragma unroll
            for (int r = 0; r < 8; ++r) {
                float v = acc[t][r] + bias_t[t];
                v = v > 0.f ? v : 0.f;            // relu(h2)
                p[r] = fmaf(v, w3_t[t], p[r]);
            }

        // butterfly sum over the 16 lanes of each half-wave
        #pragma unroll
        for (int mask = 1; mask <= 8; mask <<= 1)
            #pragma unroll
            for (int r = 0; r < 8; ++r)
                p[r] += __shfl_xor(p[r], mask, 32);

        // one lane per half writes each row: ke = relu(dot + b3); tau = ke*comp
        #pragma unroll
        for (int r = 0; r < 8; ++r) {
            if (l16 == r) {
                int rg = rbase + strip * 16 + r + 8 * half_id;
                float ke = p[r] + b3n;
                ke = ke > 0.f ? ke : 0.f;
                tau_ws[net * B + rg] = ke * comp[g * B + rg];
            }
        }
    }
}

// Sparse channel combine + trivial lw/iw outputs.
__global__ __launch_bounds__(256)
void combine_kernel(const float* __restrict__ ws,
                    const float* __restrict__ comp,
                    const float* __restrict__ b_lw,
                    const float* __restrict__ b_iw,
                    float* __restrict__ out, int B) {
    const int c = blockIdx.y;
    const int b = blockIdx.x * blockDim.x + threadIdx.x;
    if (b >= B) return;
    float s = 0.f;
    #pragma unroll
    for (int j = 0; j < 6; ++j) {
        int n = SEL[c][j];
        if (n >= 0) s += ws[n * B + b];
    }
    const int total = NCH * B;
    out[c * B + b]             = s;
    out[total + c * B + b]     = fmaxf(b_lw[c], 0.f) * comp[6 * B + b];
    out[2 * total + c * B + b] = fmaxf(b_iw[c], 0.f) * comp[7 * B + b];
}

extern "C" void kernel_launch(void* const* d_in, const int* in_sizes, int n_in,
                              void* d_out, int out_size, void* d_ws, size_t ws_size,
                              hipStream_t stream) {
    const float* t_p  = (const float*)d_in[0];
    const float* comp = (const float*)d_in[1];
    // d_in[2]/d_in[3] (null_lw / null_iw) are all-zero -> only biases matter
    const float* W1   = (const float*)d_in[4];
    const float* b1   = (const float*)d_in[5];
    const float* W2   = (const float*)d_in[6];
    const float* b2   = (const float*)d_in[7];
    const float* W3   = (const float*)d_in[8];
    const float* b3   = (const float*)d_in[9];
    const float* b_lw = (const float*)d_in[11];
    const float* b_iw = (const float*)d_in[13];
    float* out = (float*)d_out;
    float* ws  = (float*)d_ws;   // needs NNETS * B * 4 bytes (9.5 MB at B=32768)

    const int B = in_sizes[0] / 2;

    dim3 grid1(B / 256, NNETS);
    mlp_fused_kernel<<<grid1, 128, 0, stream>>>(t_p, comp, W1, b1, W2, b2, W3, b3, ws, B);

    dim3 grid2((B + 255) / 256, NCH);
    combine_kernel<<<grid2, 256, 0, stream>>>(ws, comp, b_lw, b_iw, out, B);
}
